// Summary_27178553049428
// MI455X (gfx1250) — compile-verified, matrix operand order checked by reading, not run on hardware
//
#include <hip/hip_runtime.h>
#include <hip/hip_bf16.h>

// ---------------- problem constants (match reference) ----------------
#define Dnn    128
#define Hnn    256
#define Bpairs 4096
#define NODES  8192      // 2*B
#define NBR    32

typedef __attribute__((ext_vector_type(16))) __bf16 v16bf;
typedef __attribute__((ext_vector_type(8)))  __bf16 v8bf;
typedef __attribute__((ext_vector_type(8)))  float  v8f;

static __device__ __forceinline__ __bf16 f2bf(float f) {
    unsigned u = __builtin_bit_cast(unsigned, f);
    // round-to-nearest-even bf16
    unsigned r = u + 0x7FFFu + ((u >> 16) & 1u);
    unsigned short h = (unsigned short)(r >> 16);
    return __builtin_bit_cast(__bf16, h);
}

// Load one 16x32 bf16 WMMA A/B fragment from a row-major (leading dim = ld elems)
// matrix. lane half (l>>4) selects the K sub-pattern per the CDNA5 16-bit layout:
//   dwords 0-3: K = kk + {0..7}  (half0) / kk + {8..15}  (half1)
//   dwords 4-7: K = kk + {16..23}(half0) / kk + {24..31} (half1)
// => two contiguous 16-byte chunks per lane.
static __device__ __forceinline__ v16bf load_frag(const __bf16* row, int kk, int half) {
    v8bf lo = *(const v8bf*)(row + kk + 8 * half);
    v8bf hi = *(const v8bf*)(row + kk + 16 + 8 * half);
    v16bf r;
#pragma unroll
    for (int i = 0; i < 8; ++i) { r[i] = lo[i]; r[i + 8] = hi[i]; }
    return r;
}

// ---------------- kernel 1: weight transpose/convert + bias fuse ----------------
// WtT [256][128], WnT [256][256], WrT [256][512]  (N-major bf16), bias12 = bt+bn
__global__ void prep_kernel(const float* __restrict__ Wt, const float* __restrict__ Wn,
                            const float* __restrict__ Wr, const float* __restrict__ bt,
                            const float* __restrict__ bn,
                            __bf16* __restrict__ WtT, __bf16* __restrict__ WnT,
                            __bf16* __restrict__ WrT, float* __restrict__ bias12) {
    int tid = blockIdx.x * blockDim.x + threadIdx.x;   // 0 .. 131071
    if (tid < Dnn * Hnn) {                 // 32768: WtT[n][k] = Wt[k][n]
        int n = tid / Dnn, k = tid % Dnn;
        WtT[tid] = f2bf(Wt[(size_t)k * Hnn + n]);
    }
    if (tid < 2 * Dnn * Hnn) {             // 65536: WnT[n][k] = Wn[k][n], K=256
        int n = tid / (2 * Dnn), k = tid % (2 * Dnn);
        WnT[tid] = f2bf(Wn[(size_t)k * Hnn + n]);
    }
    if (tid < 2 * Hnn * Hnn) {             // 131072: WrT[n][k] = Wr[k][n], K=512
        int n = tid / (2 * Hnn), k = tid % (2 * Hnn);
        WrT[tid] = f2bf(Wr[(size_t)k * Hnn + n]);
    }
    if (tid < Hnn) bias12[tid] = bt[tid] + bn[tid];
}

// ---------------- kernel 2: gather + 32-neighbor mean (fp32) -> bf16 activations ----
// one wave per node; lane owns float4 column slice (32 lanes * 4 = 128 cols)
__global__ void aggregate_kernel(const float* __restrict__ ent_emb,
                                 const float* __restrict__ rel_emb,
                                 const int* __restrict__ entities,
                                 const int* __restrict__ nbr_ent,
                                 const int* __restrict__ nbr_rel,
                                 __bf16* __restrict__ X1,   // [NODES][128]
                                 __bf16* __restrict__ X2) { // [NODES][256] = ne||nr
    int node = (blockIdx.x * blockDim.x + threadIdx.x) >> 5;
    int lane = threadIdx.x & 31;
    int col  = lane * 4;
    float e0 = 0.f, e1 = 0.f, e2 = 0.f, e3 = 0.f;
    float r0 = 0.f, r1 = 0.f, r2 = 0.f, r3 = 0.f;
    const int base = node * NBR;
#pragma unroll 4
    for (int j = 0; j < NBR; ++j) {
        int ie = nbr_ent[base + j];
        const float4 ve = *(const float4*)(ent_emb + (size_t)ie * Dnn + col);
        e0 += ve.x; e1 += ve.y; e2 += ve.z; e3 += ve.w;
        int ir = nbr_rel[base + j];
        const float4 vr = *(const float4*)(rel_emb + (size_t)ir * Dnn + col);
        r0 += vr.x; r1 += vr.y; r2 += vr.z; r3 += vr.w;
    }
    const float inv = 1.0f / (float)NBR;
    int se = entities[node];
    const float4 s = *(const float4*)(ent_emb + (size_t)se * Dnn + col);
    __bf16* x1 = X1 + (size_t)node * Dnn + col;
    x1[0] = f2bf(s.x); x1[1] = f2bf(s.y); x1[2] = f2bf(s.z); x1[3] = f2bf(s.w);
    __bf16* x2e = X2 + (size_t)node * (2 * Dnn) + col;
    x2e[0] = f2bf(e0 * inv); x2e[1] = f2bf(e1 * inv);
    x2e[2] = f2bf(e2 * inv); x2e[3] = f2bf(e3 * inv);
    __bf16* x2r = x2e + Dnn;
    x2r[0] = f2bf(r0 * inv); x2r[1] = f2bf(r1 * inv);
    x2r[2] = f2bf(r2 * inv); x2r[3] = f2bf(r3 * inv);
}

// ---------------- kernel 3: node = relu(X1@Wt + X2@Wn + bias12), bf16 out ----------
// block = 8 waves (2 M x 4 N); wave tile = 16 x 64 (4 accumulators of v8f)
__global__ void gemm12_kernel(const __bf16* __restrict__ X1, const __bf16* __restrict__ X2,
                              const __bf16* __restrict__ WtT, const __bf16* __restrict__ WnT,
                              const float* __restrict__ bias12,
                              __bf16* __restrict__ node) {  // [NODES][256]
    const int l    = threadIdx.x & 31;
    const int wid  = threadIdx.x >> 5;
    const int wm   = wid & 1;
    const int wn   = wid >> 1;
    const int half = l >> 4;
    const int lm   = l & 15;
    const int m0   = blockIdx.x * 32 + wm * 16;
    const int n0   = wn * 64;

    v8f acc[4] = {};

    // --- X1 [*,128] @ WtT ---
    const __bf16* arow1 = X1 + (size_t)(m0 + lm) * Dnn;
#pragma unroll
    for (int kk = 0; kk < Dnn; kk += 32) {
        v16bf a = load_frag(arow1, kk, half);
#pragma unroll
        for (int s = 0; s < 4; ++s) {
            const __bf16* brow = WtT + (size_t)(n0 + 16 * s + lm) * Dnn;
            v16bf b = load_frag(brow, kk, half);
            acc[s] = __builtin_amdgcn_wmma_f32_16x16x32_bf16(
                false, a, false, b, (short)0, acc[s], false, false);
        }
    }
    // --- X2 [*,256] @ WnT ---
    const __bf16* arow2 = X2 + (size_t)(m0 + lm) * (2 * Dnn);
#pragma unroll
    for (int kk = 0; kk < 2 * Dnn; kk += 32) {
        v16bf a = load_frag(arow2, kk, half);
#pragma unroll
        for (int s = 0; s < 4; ++s) {
            const __bf16* brow = WnT + (size_t)(n0 + 16 * s + lm) * (2 * Dnn);
            v16bf b = load_frag(brow, kk, half);
            acc[s] = __builtin_amdgcn_wmma_f32_16x16x32_bf16(
                false, a, false, b, (short)0, acc[s], false, false);
        }
    }
    // --- bias + relu + bf16 store; C layout: vgpr j -> M = j + 8*half, N = lm ---
#pragma unroll
    for (int s = 0; s < 4; ++s) {
        const int n = n0 + 16 * s + lm;
        const float bv = bias12[n];
#pragma unroll
        for (int j = 0; j < 8; ++j) {
            const int m = m0 + 8 * half + j;
            float v = fmaxf(acc[s][j] + bv, 0.0f);
            node[(size_t)m * Hnn + n] = f2bf(v);
        }
    }
}

// ---------------- kernel 4: pair = relu(node2 @ Wr + br), fp32 out -----------------
// node2 = node reinterpreted [4096][512]; WrT = [256][512]
__global__ void gemm3_kernel(const __bf16* __restrict__ node2, const __bf16* __restrict__ WrT,
                             const float* __restrict__ br, float* __restrict__ out) {
    const int l    = threadIdx.x & 31;
    const int wid  = threadIdx.x >> 5;
    const int wm   = wid & 1;
    const int wn   = wid >> 1;
    const int half = l >> 4;
    const int lm   = l & 15;
    const int m0   = blockIdx.x * 32 + wm * 16;
    const int n0   = wn * 64;

    v8f acc[4] = {};
    const __bf16* arow = node2 + (size_t)(m0 + lm) * (2 * Hnn);
#pragma unroll
    for (int kk = 0; kk < 2 * Hnn; kk += 32) {
        v16bf a = load_frag(arow, kk, half);
#pragma unroll
        for (int s = 0; s < 4; ++s) {
            const __bf16* brow = WrT + (size_t)(n0 + 16 * s + lm) * (2 * Hnn);
            v16bf b = load_frag(brow, kk, half);
            acc[s] = __builtin_amdgcn_wmma_f32_16x16x32_bf16(
                false, a, false, b, (short)0, acc[s], false, false);
        }
    }
#pragma unroll
    for (int s = 0; s < 4; ++s) {
        const int n = n0 + 16 * s + lm;
        const float bv = br[n];
#pragma unroll
        for (int j = 0; j < 8; ++j) {
            const int m = m0 + 8 * half + j;
            out[(size_t)m * Hnn + n] = fmaxf(acc[s][j] + bv, 0.0f);
        }
    }
}

// ---------------- host-side launch ----------------
extern "C" void kernel_launch(void* const* d_in, const int* in_sizes, int n_in,
                              void* d_out, int out_size, void* d_ws, size_t ws_size,
                              hipStream_t stream) {
    const float* ent_emb = (const float*)d_in[0];   // [500000,128]
    const float* rel_emb = (const float*)d_in[1];   // [500,128]
    const float* Wt      = (const float*)d_in[2];   // [128,256]
    const float* bt      = (const float*)d_in[3];   // [256]
    const float* Wn      = (const float*)d_in[4];   // [256,256]
    const float* bn      = (const float*)d_in[5];   // [256]
    const float* Wr      = (const float*)d_in[6];   // [512,256]
    const float* br      = (const float*)d_in[7];   // [256]
    const int* entities  = (const int*)d_in[8];     // [8192]
    const int* nbr_ent   = (const int*)d_in[9];     // [262144]
    const int* nbr_rel   = (const int*)d_in[10];    // [262144]
    // d_in[11] = offsets: uniform stride NBR=32, folded into indexing.

    // workspace carve-up (256B aligned chunks)
    char* ws = (char*)d_ws;
    size_t off = 0;
    auto take = [&](size_t bytes) { char* p = ws + off; off = (off + bytes + 255) & ~(size_t)255; return p; };
    __bf16* X1     = (__bf16*)take((size_t)NODES * Dnn * 2);       // 2 MB
    __bf16* X2     = (__bf16*)take((size_t)NODES * 2 * Dnn * 2);   // 4 MB
    __bf16* node   = (__bf16*)take((size_t)NODES * Hnn * 2);       // 4 MB
    __bf16* WtT    = (__bf16*)take((size_t)Dnn * Hnn * 2);
    __bf16* WnT    = (__bf16*)take((size_t)2 * Dnn * Hnn * 2);
    __bf16* WrT    = (__bf16*)take((size_t)2 * Hnn * Hnn * 2);
    float*  bias12 = (float*)take((size_t)Hnn * 4);

    prep_kernel<<<(2 * Hnn * Hnn) / 256, 256, 0, stream>>>(Wt, Wn, Wr, bt, bn,
                                                           WtT, WnT, WrT, bias12);
    aggregate_kernel<<<(NODES * 32) / 256, 256, 0, stream>>>(ent_emb, rel_emb, entities,
                                                             nbr_ent, nbr_rel, X1, X2);
    gemm12_kernel<<<NODES / 32, 256, 0, stream>>>(X1, X2, WtT, WnT, bias12, node);
    gemm3_kernel<<<Bpairs / 32, 256, 0, stream>>>(node, WrT, br, (float*)d_out);
}
